// Transformer_3736621548277
// MI455X (gfx1250) — matrix operand order repaired and checked by reading, not compiled, not run on hardware
//
#include <hip/hip_runtime.h>
#include <hip/hip_bf16.h>
#include <cstddef>

// ---- problem constants (match reference) ----
#define B_   4
#define S_   512
#define D_   512
#define H_   8
#define DK_  64
#define NL_  6
#define V_   32000
#define FF_  2048
#define BS_  (B_ * S_)
#define DD_  (D_ * D_)

typedef __bf16 bf16;
typedef bf16  v4bf  __attribute__((ext_vector_type(4)));
typedef bf16  v16bf __attribute__((ext_vector_type(16)));
typedef float v8f   __attribute__((ext_vector_type(8)));

#define KSTEP 64   // two 32-wide WMMA K-chunks per LDS stage

// =====================================================================
// Batched GEMM:  C[M,N] = alpha * (A[M,K] x B[K,N]) (+bias) (+relu)
// flags: 1 = B accessed transposed (B[n*ldb + k]), 2 = relu, 4 = bias
// grid = (N/32, M/64, batch); block = 128 (4 waves).
// Each wave computes two 16x16 tiles (shared A fragment); 4 WMMAs per
// K-stage. Staging is phase-split: batched global_load_b128 into regs,
// barrier, packed bf16 convert + ds_store_b64, barrier, fragments+WMMA.
// A/B fragments use contiguous k order (k = 16h + s), not the ISA table; LDS stores k
// unpermuted (per 32-chunk) so each lane reads one contiguous
// v16bf (2 x ds_read_b128).
// =====================================================================
__global__ void __launch_bounds__(128)
gemm_wmma_bf16(const float* __restrict__ A, const float* __restrict__ Bm,
               const float* __restrict__ bias, float* __restrict__ C,
               int M, int N, int K, int lda, int ldb, int ldc,
               int flags, float alpha, int Hdiv,
               long long sAb, long long sAh, long long sBb, long long sBh,
               long long sCb, long long sCh)
{
    __shared__ bf16 As[64 * KSTEP];   // 8 KB
    __shared__ bf16 Bt[32 * KSTEP];   // 4 KB, Bt[n][k]

    const int t    = threadIdx.x;
    const int lane = t & 31;
    const int wave = t >> 5;
    const int tileN = blockIdx.x << 5;   // 32 columns per block
    const int tileM = blockIdx.y << 6;   // 64 rows per block
    const int z  = blockIdx.z;
    const int zb = z / Hdiv, zh = z - zb * Hdiv;
    A  += zb * sAb + zh * sAh;
    Bm += zb * sBb + zh * sBh;
    C  += zb * sCb + zh * sCh;

    v8f acc0 = {};
    v8f acc1 = {};

    for (int k0 = 0; k0 < K; k0 += KSTEP) {
        // ---- phase 1: batched global loads into registers (b128) ----
        float4 ra[8];                        // A: 64x64 floats / 128 thr
#pragma unroll
        for (int i = 0; i < 8; ++i) {
            int idx4 = (i << 7) + t;         // 0..1023 float4s
            int m = idx4 >> 4, kq = (idx4 & 15) << 2;
            ra[i] = *(const float4*)&A[(long long)(tileM + m) * lda + (k0 + kq)];
        }
        float4 rb[4];                        // B: 32x64 floats / 128 thr
        if (flags & 1) {                     // B given as [n][k]
#pragma unroll
            for (int i = 0; i < 4; ++i) {
                int idx4 = (i << 7) + t;     // 0..511
                int n = idx4 >> 4, kq = (idx4 & 15) << 2;
                rb[i] = *(const float4*)&Bm[(long long)(tileN + n) * ldb + (k0 + kq)];
            }
        } else {                             // B given as [k][n]
#pragma unroll
            for (int i = 0; i < 4; ++i) {
                int idx4 = (i << 7) + t;
                int kk = idx4 >> 3, nq = (idx4 & 7) << 2;
                rb[i] = *(const float4*)&Bm[(long long)(k0 + kk) * ldb + (tileN + nq)];
            }
        }
        if (k0 + KSTEP < K)                  // pull next A tile toward L2
            __builtin_prefetch(&A[(long long)(tileM + (t >> 1)) * lda + (k0 + KSTEP)], 0, 1);

        __syncthreads();                     // prior LDS reads complete

        // ---- phase 2: packed convert + vector LDS stores ----
#pragma unroll
        for (int i = 0; i < 8; ++i) {
            int idx4 = (i << 7) + t;
            int m = idx4 >> 4, kq = (idx4 & 15) << 2;
            int ks = kq;
            v4bf pk;
            pk[0] = (bf16)ra[i].x; pk[1] = (bf16)ra[i].y;
            pk[2] = (bf16)ra[i].z; pk[3] = (bf16)ra[i].w;
            *(v4bf*)&As[m * KSTEP + ks] = pk;     // ds_store_b64
        }
        if (flags & 1) {
#pragma unroll
            for (int i = 0; i < 4; ++i) {
                int idx4 = (i << 7) + t;
                int n = idx4 >> 4, kq = (idx4 & 15) << 2;
                v4bf pk;
                pk[0] = (bf16)rb[i].x; pk[1] = (bf16)rb[i].y;
                pk[2] = (bf16)rb[i].z; pk[3] = (bf16)rb[i].w;
                *(v4bf*)&Bt[n * KSTEP + kq] = pk;
            }
        } else {                             // transpose into LDS
#pragma unroll
            for (int i = 0; i < 4; ++i) {
                int idx4 = (i << 7) + t;
                int kk = idx4 >> 3, nq = (idx4 & 7) << 2;
                Bt[(nq + 0) * KSTEP + kk] = (bf16)rb[i].x;
                Bt[(nq + 1) * KSTEP + kk] = (bf16)rb[i].y;
                Bt[(nq + 2) * KSTEP + kk] = (bf16)rb[i].z;
                Bt[(nq + 3) * KSTEP + kk] = (bf16)rb[i].w;
            }
        }
        __syncthreads();

        // ---- phase 3: fragments + 4 WMMAs ----
        const int fm = (wave << 4) + (lane & 15);
        const int g  = lane & 16;            // K-half select (contiguous k = 16h + s)
#pragma unroll
        for (int c = 0; c < 2; ++c) {
            v16bf a  = *(const v16bf*)&As[fm * KSTEP + c * 32 + g];
            v16bf b0 = *(const v16bf*)&Bt[(lane & 15) * KSTEP + c * 32 + g];
            v16bf b1 = *(const v16bf*)&Bt[((lane & 15) + 16) * KSTEP + c * 32 + g];
            acc0 = __builtin_amdgcn_wmma_f32_16x16x32_bf16(false, a, false, b0,
                                                           (short)0, acc0, false, false);
            acc1 = __builtin_amdgcn_wmma_f32_16x16x32_bf16(false, a, false, b1,
                                                           (short)0, acc1, false, false);
        }
    }

    // ---- store (C/D layout: VGPR r -> M = r + 8*(lane>=16), N = lane&15) ----
    const int row0 = tileM + (wave << 4) + ((lane & 16) >> 1);
    const int col0 = tileN + (lane & 15);
    float bv0 = (flags & 4) ? bias[col0]      : 0.0f;
    float bv1 = (flags & 4) ? bias[col0 + 16] : 0.0f;
#pragma unroll
    for (int r = 0; r < 8; ++r) {
        float v0 = acc0[r] * alpha + bv0;
        float v1 = acc1[r] * alpha + bv1;
        if (flags & 2) { v0 = fmaxf(v0, 0.0f); v1 = fmaxf(v1, 0.0f); }
        C[(long long)(row0 + r) * ldc + col0]      = v0;
        C[(long long)(row0 + r) * ldc + col0 + 16] = v1;
    }
}

// =====================================================================
// Embedding gather + sinusoidal positional encoding (faithful to ref:
// ang = pos / 10000^(2j/D); even j -> sin, odd j -> cos)
// =====================================================================
__global__ void __launch_bounds__(256)
embed_pe(const int* __restrict__ tok, const float* __restrict__ emb,
         float* __restrict__ out)
{
    int idx = blockIdx.x * 256 + threadIdx.x;  // over B*S*D
    int d  = idx & (D_ - 1);
    int bs = idx >> 9;           // D_ = 512
    int s  = bs & (S_ - 1);
    float inv = __expf(-(2.0f * (float)d / (float)D_) * 9.2103403719761836f); // ln 1e4
    float ang = (float)s * inv;
    float pe  = (d & 1) ? __cosf(ang) : __sinf(ang);
    out[idx] = emb[(long long)tok[bs] * D_ + d] + pe;
}

// =====================================================================
// out = LayerNorm(a + b)  over last dim (D=512). One block per row.
// Safe for out == b (all reads of b precede the first barrier).
// =====================================================================
__global__ void __launch_bounds__(256)
ln_residual(const float* __restrict__ a, const float* __restrict__ b,
            float* __restrict__ out)
{
    const int row = blockIdx.x, t = threadIdx.x;
    const float* pa = a + (long long)row * D_;
    const float* pb = b + (long long)row * D_;
    float v0 = pa[t] + pb[t];
    float v1 = pa[t + 256] + pb[t + 256];

    __shared__ float buf[256];
    buf[t] = v0 + v1; __syncthreads();
    for (int s = 128; s; s >>= 1) { if (t < s) buf[t] += buf[t + s]; __syncthreads(); }
    float mean = buf[0] * (1.0f / D_);
    __syncthreads();
    float d0 = v0 - mean, d1 = v1 - mean;
    buf[t] = d0 * d0 + d1 * d1; __syncthreads();
    for (int s = 128; s; s >>= 1) { if (t < s) buf[t] += buf[t + s]; __syncthreads(); }
    float rstd = __frsqrt_rn(buf[0] * (1.0f / D_) + 1e-5f);

    float* po = out + (long long)row * D_;
    po[t]       = d0 * rstd;
    po[t + 256] = d1 * rstd;
}

// =====================================================================
// In-place row softmax; causal!=0 masks columns c > (row % S) to -1e9
// (matches jnp.where(mask, s, -1e9) then softmax). One block per row.
// =====================================================================
__global__ void __launch_bounds__(256)
softmax_inplace(float* __restrict__ x, long long ncols, int causal)
{
    const long long row = blockIdx.x;
    float* p = x + row * ncols;
    const int q = causal ? (int)(row & (S_ - 1)) : 0x7fffffff;
    const int t = threadIdx.x;
    __shared__ float buf[256];

    float mx = -3.4e38f;
    for (int c = t; c < (int)ncols; c += 256) {
        float v = p[c];
        if (c > q) v = -1e9f;
        mx = fmaxf(mx, v);
    }
    buf[t] = mx; __syncthreads();
    for (int s = 128; s; s >>= 1) { if (t < s) buf[t] = fmaxf(buf[t], buf[t + s]); __syncthreads(); }
    mx = buf[0]; __syncthreads();

    float sum = 0.0f;
    for (int c = t; c < (int)ncols; c += 256) {
        float v = p[c];
        if (c > q) v = -1e9f;
        v = __expf(v - mx);
        p[c] = v;
        sum += v;
    }
    buf[t] = sum; __syncthreads();
    for (int s = 128; s; s >>= 1) { if (t < s) buf[t] += buf[t + s]; __syncthreads(); }
    float invs = 1.0f / buf[0];
    __syncthreads();
    for (int c = t; c < (int)ncols; c += 256) p[c] *= invs;
}

// =====================================================================
// Host-side orchestration
// =====================================================================
static inline void launch_gemm(hipStream_t st, const float* A, const float* Bm,
                               const float* bias, float* C,
                               int M, int N, int K, int lda, int ldb, int ldc,
                               int flags, float alpha,
                               int batch = 1, int Hdiv = 1,
                               long long sAb = 0, long long sAh = 0,
                               long long sBb = 0, long long sBh = 0,
                               long long sCb = 0, long long sCh = 0)
{
    dim3 grid(N / 32, M / 64, batch);
    gemm_wmma_bf16<<<grid, 128, 0, st>>>(A, Bm, bias, C, M, N, K, lda, ldb, ldc,
                                         flags, alpha, Hdiv,
                                         sAb, sAh, sBb, sBh, sCb, sCh);
}

struct Bufs {
    float *x, *y, *q, *k, *v, *ao, *mo, *hh, *sc;
};

static void mha(hipStream_t st, const float* xq, const float* xkv, const float* w4,
                Bufs& ws, float* out, int causal)
{
    const float* wq = w4;
    const float* wk = w4 + DD_;
    const float* wv = w4 + 2 * DD_;
    const float* wo = w4 + 3 * DD_;
    // projections: [B*S, D] @ [D, D]
    launch_gemm(st, xq,  wq, nullptr, ws.q, BS_, D_, D_, D_, D_, D_, 0, 1.0f);
    launch_gemm(st, xkv, wk, nullptr, ws.k, BS_, D_, D_, D_, D_, D_, 0, 1.0f);
    launch_gemm(st, xkv, wv, nullptr, ws.v, BS_, D_, D_, D_, D_, D_, 0, 1.0f);
    // scores[b,h,q,k] = (Q . K^T) / sqrt(DK); batched over z = b*H + h
    launch_gemm(st, ws.q, ws.k, nullptr, ws.sc, S_, S_, DK_, D_, D_, S_,
                /*flags=Btrans*/1, 0.125f, B_ * H_, H_,
                (long long)S_ * D_, DK_,              // A: q per (b,h)
                (long long)S_ * D_, DK_,              // B: k per (b,h)
                (long long)H_ * S_ * S_, (long long)S_ * S_); // C
    softmax_inplace<<<B_ * H_ * S_, 256, 0, st>>>(ws.sc, (long long)S_, causal);
    // attn @ V -> concat heads
    launch_gemm(st, ws.sc, ws.v, nullptr, ws.ao, S_, DK_, S_, S_, D_, D_,
                0, 1.0f, B_ * H_, H_,
                (long long)H_ * S_ * S_, (long long)S_ * S_,
                (long long)S_ * D_, DK_,
                (long long)S_ * D_, DK_);
    // output projection
    launch_gemm(st, ws.ao, wo, nullptr, out, BS_, D_, D_, D_, D_, D_, 0, 1.0f);
}

extern "C" void kernel_launch(void* const* d_in, const int* in_sizes, int n_in,
                              void* d_out, int out_size, void* d_ws, size_t ws_size,
                              hipStream_t stream)
{
    (void)in_sizes; (void)n_in; (void)out_size; (void)ws_size;
    const int*   src      = (const int*)  d_in[0];
    const int*   trg      = (const int*)  d_in[1];
    const float* src_emb  = (const float*)d_in[2];
    const float* trg_emb  = (const float*)d_in[3];
    const float* enc_qkvo = (const float*)d_in[4];
    const float* e_w1     = (const float*)d_in[5];
    const float* e_b1     = (const float*)d_in[6];
    const float* e_w2     = (const float*)d_in[7];
    const float* e_b2     = (const float*)d_in[8];
    const float* dsq      = (const float*)d_in[9];
    const float* dcq      = (const float*)d_in[10];
    const float* d_w1     = (const float*)d_in[11];
    const float* d_b1     = (const float*)d_in[12];
    const float* d_w2     = (const float*)d_in[13];
    const float* d_b2     = (const float*)d_in[14];
    const float* out_w    = (const float*)d_in[15];
    const float* out_b    = (const float*)d_in[16];
    float* out = (float*)d_out;

    // ---- carve workspace (all buffers fully written before read) ----
    float* p = (float*)d_ws;
    Bufs ws;
    ws.x  = p; p += (size_t)BS_ * D_;
    ws.y  = p; p += (size_t)BS_ * D_;
    ws.q  = p; p += (size_t)BS_ * D_;
    ws.k  = p; p += (size_t)BS_ * D_;
    ws.v  = p; p += (size_t)BS_ * D_;
    ws.ao = p; p += (size_t)BS_ * D_;
    ws.mo = p; p += (size_t)BS_ * D_;
    ws.hh = p; p += (size_t)BS_ * FF_;
    ws.sc = p; p += (size_t)B_ * H_ * S_ * S_;

    const int embBlocks = (B_ * S_ * D_) / 256;
    embed_pe<<<embBlocks, 256, 0, stream>>>(src, src_emb, ws.x);
    embed_pe<<<embBlocks, 256, 0, stream>>>(trg, trg_emb, ws.y);

    // ---------------- encoder ----------------
    for (int i = 0; i < NL_; ++i) {
        mha(stream, ws.x, ws.x, enc_qkvo + (size_t)i * 4 * DD_, ws, ws.mo, 0);
        ln_residual<<<BS_, 256, 0, stream>>>(ws.mo, ws.x, ws.x);
        launch_gemm(stream, ws.x, e_w1 + (size_t)i * D_ * FF_, e_b1 + (size_t)i * FF_,
                    ws.hh, BS_, FF_, D_, D_, FF_, FF_, 4 | 2, 1.0f);
        launch_gemm(stream, ws.hh, e_w2 + (size_t)i * FF_ * D_, e_b2 + (size_t)i * D_,
                    ws.mo, BS_, D_, FF_, FF_, D_, D_, 4, 1.0f);
        ln_residual<<<BS_, 256, 0, stream>>>(ws.mo, ws.x, ws.x);
    }

    // ---------------- decoder (cross-attends to final encoder output) ----------------
    for (int i = 0; i < NL_; ++i) {
        mha(stream, ws.y, ws.y, dsq + (size_t)i * 4 * DD_, ws, ws.mo, /*causal=*/1);
        ln_residual<<<BS_, 256, 0, stream>>>(ws.mo, ws.y, ws.y);
        mha(stream, ws.y, ws.x, dcq + (size_t)i * 4 * DD_, ws, ws.mo, 0);
        ln_residual<<<BS_, 256, 0, stream>>>(ws.mo, ws.y, ws.y);
        launch_gemm(stream, ws.y, d_w1 + (size_t)i * D_ * FF_, d_b1 + (size_t)i * FF_,
                    ws.hh, BS_, FF_, D_, D_, FF_, FF_, 4 | 2, 1.0f);
        launch_gemm(stream, ws.hh, d_w2 + (size_t)i * FF_ * D_, d_b2 + (size_t)i * D_,
                    ws.mo, BS_, D_, FF_, FF_, D_, D_, 4, 1.0f);
        ln_residual<<<BS_, 256, 0, stream>>>(ws.mo, ws.y, ws.y);
    }

    // ---------------- output head: logits into d_out, softmax in place ----------------
    launch_gemm(stream, ws.y, out_w, out_b, out, BS_, V_, D_, D_, V_, V_, 4, 1.0f);
    softmax_inplace<<<BS_, 256, 0, stream>>>(out, (long long)V_, 0);
}
